// CustomLSTM_38190849196289
// MI455X (gfx1250) — matrix operand order repaired and checked by reading, not compiled, not run on hardware
//
#include <hip/hip_runtime.h>
#include <hip/hip_bf16.h>

// ---------------------------------------------------------------------------
// CustomLSTM on gfx1250 (MI455X): B=64, S=512, I=256, H=512 (4H=2048)
//   prep:       f32->bf16 convert of x; transpose+convert W,U to [N][K] bf16
//   xproj_gemm: x_proj[t][b][n] = x@W + bias  (bf16 WMMA, f32 accum)
//   lstm_rec:   persistent 32-WG kernel, 512 steps, register-resident U,
//               1 grid barrier per step, c-state in registers
// ---------------------------------------------------------------------------

#define BB 64
#define SS 512
#define II 256
#define HH 512
#define N4 2048
#define NWG_REC 32

typedef __attribute__((ext_vector_type(16))) __bf16 v16bf;
typedef __attribute__((ext_vector_type(8)))  float  v8f;

union Frag {            // 16 bf16 = 32 bytes = two 16B global loads
    v16bf v;
    uint4 q[2];
};

__device__ inline unsigned short f2bf(float f) {
    unsigned u = __float_as_uint(f);
    u += 0x7FFFu + ((u >> 16) & 1u);     // round-to-nearest-even
    return (unsigned short)(u >> 16);
}
__device__ inline float sigm(float x)  { return 1.0f / (1.0f + __expf(-x)); }
__device__ inline float tanhf_(float x){ float e = __expf(-2.0f * x); return (1.0f - e) / (1.0f + e); }

// Monotonic-counter grid barrier (32 resident WGs).
__device__ inline void grid_barrier(unsigned* cnt, unsigned target) {
    __threadfence();
    __syncthreads();
    if (threadIdx.x == 0) {
        __hip_atomic_fetch_add(cnt, 1u, __ATOMIC_ACQ_REL, __HIP_MEMORY_SCOPE_AGENT);
        while (__hip_atomic_load(cnt, __ATOMIC_ACQUIRE, __HIP_MEMORY_SCOPE_AGENT) < target)
            __builtin_amdgcn_s_sleep(2);
    }
    __syncthreads();
    __threadfence();
}

// ---------------------------------------------------------------------------
// prep: convert/transpose weights + activations, zero barrier counter
// ---------------------------------------------------------------------------
__global__ void lstm_prep(const float* __restrict__ x, const float* __restrict__ W,
                          const float* __restrict__ U,
                          unsigned short* __restrict__ xb,   // [B*S][I] bf16
                          unsigned short* __restrict__ Wt,   // [N4][I]  bf16 (transposed)
                          unsigned short* __restrict__ Ut,   // [N4][H]  bf16 (transposed)
                          unsigned* cnt) {
    int j = blockIdx.x * blockDim.x + threadIdx.x;
    if (j == 0) *cnt = 0;
    const int NX = BB * SS * II;
    if (j < NX) xb[j] = f2bf(x[j]);
    const int NW = II * N4;
    if (j < NW) { int n = j / II, k = j % II; Wt[j] = f2bf(W[(size_t)k * N4 + n]); }
    const int NU = HH * N4;
    if (j < NU) { int n = j / HH, k = j % HH; Ut[j] = f2bf(U[(size_t)k * N4 + n]); }
}

// ---------------------------------------------------------------------------
// xproj_gemm: M=32768 (b*S+s), K=256, N=2048. One wave = 64x64 tile.
// K loop NOT unrolled: keeps VGPRs < 256 (no MSB juggling), better occupancy
// for this HBM-bound phase.
// Output layout is recurrence-friendly: xproj[((s*64)+b)*2048 + n].
// ---------------------------------------------------------------------------
__global__ __launch_bounds__(128) void xproj_gemm(const unsigned short* __restrict__ xb,
                                                  const unsigned short* __restrict__ Wt,
                                                  const float* __restrict__ bias,
                                                  float* __restrict__ xproj) {
    const int wave  = (blockIdx.x * blockDim.x + threadIdx.x) >> 5;
    const int lane  = threadIdx.x & 31;
    const int lhalf = lane >> 4, l15 = lane & 15;
    const int tm = wave >> 5;          // 512 M-tiles of 64
    const int tn = wave & 31;          // 32  N-tiles of 64
    const int m_base = tm * 64, n_base = tn * 64;

    v8f acc[4][4] = {};
#pragma unroll 1
    for (int kc = 0; kc < II; kc += 32) {
        Frag a[4], b[4];
#pragma unroll
        for (int i = 0; i < 4; i++) {
            const int m  = m_base + 16 * i + l15;
            const int ka = kc + lhalf * 8;                       // A: K {ka..+7, ka+16..+23}
            a[i].q[0] = *(const uint4*)(xb + (size_t)m * II + ka);
            a[i].q[1] = *(const uint4*)(xb + (size_t)m * II + ka + 16);
            const int n  = n_base + 16 * i + l15;
            const int kb = kc + lhalf * 8;                       // B: K {kb..+7, kb+16..+23}
            b[i].q[0] = *(const uint4*)(Wt + (size_t)n * II + kb);
            b[i].q[1] = *(const uint4*)(Wt + (size_t)n * II + kb + 16);
        }
#pragma unroll
        for (int i = 0; i < 4; i++)
#pragma unroll
            for (int j = 0; j < 4; j++)
                acc[i][j] = __builtin_amdgcn_wmma_f32_16x16x32_bf16(
                    false, a[i].v, false, b[j].v, (short)0, acc[i][j], false, false);
    }
#pragma unroll
    for (int j = 0; j < 4; j++) {
        const int n = n_base + 16 * j + l15;
        const float bs = bias[n];
#pragma unroll
        for (int i = 0; i < 4; i++) {
#pragma unroll
            for (int r = 0; r < 8; r++) {
                const int m = m_base + 16 * i + 8 * lhalf + r;   // = b*S + s
                const int b_ = m >> 9, s_ = m & 511;
                xproj[((size_t)(s_ * BB + b_)) * N4 + n] = acc[i][j][r] + bs;
            }
        }
    }
}

// ---------------------------------------------------------------------------
// Persistent recurrence. 32 WGs x 512 threads (16 waves).
// WG wg owns h-columns [16*wg, 16*wg+16). Wave (g,mt): gate tile
// rows 16*mt..+15 (batch), cols g*512 + 16*wg (i/f/g/o). U slice lives in
// 16 v16bf register fragments per wave for all 512 steps. Two independent
// accumulator chains halve the WMMA dependency latency per step.
// ---------------------------------------------------------------------------
__global__ __launch_bounds__(512) void lstm_rec(const unsigned short* __restrict__ Ut,
                                                const float* __restrict__ xproj,
                                                unsigned short* __restrict__ hb0,
                                                unsigned short* __restrict__ hb1,
                                                float* __restrict__ out_seq,
                                                float* __restrict__ out_hf,
                                                float* __restrict__ out_cf,
                                                unsigned* cnt) {
    __shared__ float lds_g[4][4][16][16];     // [gate][m_tile][row][col] = 16 KB

    const int tid   = threadIdx.x;
    const int wg    = blockIdx.x;             // 0..31
    const int wave  = tid >> 5, lane = tid & 31;
    const int g     = wave >> 2, mt = wave & 3;
    const int lhalf = lane >> 4, l15 = lane & 15;
    const int hcol0 = wg * 16;
    const int n0    = g * HH + hcol0;         // gate column base in [0,2048)

    // --- preload this wave's U slice into registers (16 cols x 512 K bf16) ---
    Frag breg[16];
    {
        const int n = n0 + l15;
#pragma unroll
        for (int kc = 0; kc < 16; kc++) {
            const int kb = kc * 32 + lhalf * 8;
            breg[kc].q[0] = *(const uint4*)(Ut + (size_t)n * HH + kb);
            breg[kc].q[1] = *(const uint4*)(Ut + (size_t)n * HH + kb + 16);
        }
    }

    // --- zero both h ping-pong buffers (16384 dwords each; 16384 threads) ---
    {
        const int idx = wg * 512 + tid;
        ((unsigned*)hb0)[idx] = 0u;
        ((unsigned*)hb1)[idx] = 0u;
    }

    // elementwise ownership: 2 consecutive (b,col) elements per thread
    const int e0   = tid * 2;
    const int row0 = e0 >> 4;                 // batch b, 0..63
    const int col0 = e0 & 15;                 // col0 even; also handles col0+1
    float cst0 = 0.0f, cst1 = 0.0f;           // persistent cell state

    grid_barrier(cnt, NWG_REC);               // h buffers zeroed everywhere

#pragma unroll 1
    for (int t = 0; t < SS; t++) {
        const unsigned short* hin  = (t & 1) ? hb1 : hb0;
        unsigned short*       hout = (t & 1) ? hb0 : hb1;

        // acc0 = x_proj[t] tile (already includes bias); acc1 = 0
        v8f acc0, acc1 = {};
        {
            const int n  = n0 + l15;
            const int b0 = mt * 16 + 8 * lhalf;
#pragma unroll
            for (int r = 0; r < 8; r++)
                acc0[r] = xproj[((size_t)(t * BB + b0 + r)) * N4 + n];
        }
        // gates += h_t @ U   (K = 512, 16 WMMAs, B from registers, 2 chains)
#pragma unroll
        for (int kc = 0; kc < 16; kc += 2) {
            Frag a0, a1;
            const int m  = mt * 16 + l15;                // batch row
            const int ka = kc * 32 + lhalf * 8;
            a0.q[0] = *(const uint4*)(hin + (size_t)m * HH + ka);
            a0.q[1] = *(const uint4*)(hin + (size_t)m * HH + ka + 16);
            a1.q[0] = *(const uint4*)(hin + (size_t)m * HH + ka + 32);
            a1.q[1] = *(const uint4*)(hin + (size_t)m * HH + ka + 48);
            acc0 = __builtin_amdgcn_wmma_f32_16x16x32_bf16(
                false, a0.v, false, breg[kc].v, (short)0, acc0, false, false);
            acc1 = __builtin_amdgcn_wmma_f32_16x16x32_bf16(
                false, a1.v, false, breg[kc + 1].v, (short)0, acc1, false, false);
        }
        acc0 = acc0 + acc1;

        // share gate tiles inside the WG
#pragma unroll
        for (int r = 0; r < 8; r++)
            lds_g[g][mt][8 * lhalf + r][l15] = acc0[r];
        __syncthreads();

        // elementwise LSTM cell for 2 owned (b, col) elements
        {
            const int rt = row0 >> 4, rr = row0 & 15;
            float xi0 = lds_g[0][rt][rr][col0],     xi1 = lds_g[0][rt][rr][col0 + 1];
            float xf0 = lds_g[1][rt][rr][col0],     xf1 = lds_g[1][rt][rr][col0 + 1];
            float xg0 = lds_g[2][rt][rr][col0],     xg1 = lds_g[2][rt][rr][col0 + 1];
            float xo0 = lds_g[3][rt][rr][col0],     xo1 = lds_g[3][rt][rr][col0 + 1];

            cst0 = sigm(xf0) * cst0 + sigm(xi0) * tanhf_(xg0);
            cst1 = sigm(xf1) * cst1 + sigm(xi1) * tanhf_(xg1);
            const float h0 = sigm(xo0) * tanhf_(cst0);
            const float h1 = sigm(xo1) * tanhf_(cst1);

            const size_t so = ((size_t)row0 * SS + t) * HH + hcol0 + col0;
            out_seq[so]     = h0;
            out_seq[so + 1] = h1;
            const size_t ho = (size_t)row0 * HH + hcol0 + col0;
            hout[ho]     = f2bf(h0);
            hout[ho + 1] = f2bf(h1);
            if (t == SS - 1) {
                out_hf[ho] = h0; out_hf[ho + 1] = h1;
                out_cf[ho] = cst0; out_cf[ho + 1] = cst1;
            }
        }
        grid_barrier(cnt, (unsigned)(NWG_REC * (t + 2)));
    }
}

// ---------------------------------------------------------------------------
extern "C" void kernel_launch(void* const* d_in, const int* in_sizes, int n_in,
                              void* d_out, int out_size, void* d_ws, size_t ws_size,
                              hipStream_t stream) {
    const float* x    = (const float*)d_in[0];
    const float* W    = (const float*)d_in[1];
    const float* U    = (const float*)d_in[2];
    const float* bias = (const float*)d_in[3];

    float* out_seq = (float*)d_out;                         // [B][S][H]
    float* out_hf  = out_seq + (size_t)BB * SS * HH;        // [B][H]
    float* out_cf  = out_hf + (size_t)BB * HH;              // [B][H]

    char* ws = (char*)d_ws;
    unsigned*       cnt = (unsigned*)ws;                    // barrier counter
    unsigned short* xb  = (unsigned short*)(ws + 256);                      // 16 MB
    unsigned short* Wt  = xb + (size_t)BB * SS * II;                        // 1 MB
    unsigned short* Ut  = Wt + (size_t)N4 * II;                             // 2 MB
    unsigned short* hb0 = Ut + (size_t)N4 * HH;                             // 64 KB
    unsigned short* hb1 = hb0 + (size_t)BB * HH;                            // 64 KB
    float*        xproj = (float*)(hb1 + (size_t)BB * HH);                  // 256 MB

    const int NX = BB * SS * II;                            // 8.4M, largest prep count
    lstm_prep<<<(NX + 255) / 256, 256, 0, stream>>>(x, W, U, xb, Wt, Ut, cnt);

    // 16384 waves of 64x64 tiles: (32768/64)*(2048/64) tiles, 4 waves/WG
    xproj_gemm<<<16384 / 4, 128, 0, stream>>>(xb, Wt, bias, xproj);

    lstm_rec<<<NWG_REC, 512, 0, stream>>>(Ut, xproj, hb0, hb1,
                                          out_seq, out_hf, out_cf, cnt);
}
